// GCN_43679817400704
// MI455X (gfx1250) — compile-verified, matrix operand order checked by reading, not run on hardware
//
#include <hip/hip_runtime.h>
#include <hip/hip_bf16.h>

// ---------------------------------------------------------------------------
// GCN forward for MI455X (gfx1250, wave32).
//  - Edge scatter: 16 threads/edge, b128 message loads, f32 atomics into an
//    L2-resident accumulator (12.8 MB << 192 MB L2).
//  - Dense 64x64 layers: v_wmma_f32_16x16x32_f16, with the self-loop + bias
//    accumulator init fused into the WMMA epilogue.
// ---------------------------------------------------------------------------

#define HID 64

typedef __attribute__((ext_vector_type(16))) _Float16 v16h;
typedef __attribute__((ext_vector_type(8)))  float    v8f;

union V16HU { v16h v; uint4 q[2]; };
union H4U   { _Float16 h[4]; uint2 u; };

// ---- degree / normalization ------------------------------------------------
__global__ void k_fill(float* p, float v, int n) {
    int i = blockIdx.x * blockDim.x + threadIdx.x;
    if (i < n) p[i] = v;
}

__global__ void k_deg_acc(const int* __restrict__ dst, float* __restrict__ deg, int e) {
    int i = blockIdx.x * blockDim.x + threadIdx.x;
    if (i < e) atomicAdd(&deg[dst[i]], 1.0f);
}

__global__ void k_rsqrt_inplace(float* p, int n) {
    int i = blockIdx.x * blockDim.x + threadIdx.x;
    if (i < n) p[i] = rsqrtf(p[i]);   // deg >= 1 always (self-loops)
}

__global__ void k_edge_norm(const int* __restrict__ src, const int* __restrict__ dst,
                            const float* __restrict__ dinv, float* __restrict__ norm, int e) {
    int i = blockIdx.x * blockDim.x + threadIdx.x;
    if (i < e) norm[i] = dinv[src[i]] * dinv[dst[i]];
}

// ---- pack W (K x Nn, row-major f32) into WMMA B lane layout ----------------
__global__ void k_pack_w(const float* __restrict__ W, _Float16* __restrict__ P,
                         int K, int Nn) {
    int ktiles = K / 32;
    int total  = (Nn / 16) * ktiles * 32 * 16;
    int idx = blockIdx.x * blockDim.x + threadIdx.x;
    if (idx >= total) return;
    int s    = idx & 15;
    int lane = (idx >> 4) & 31;
    int kt   = (idx >> 9) % ktiles;
    int nt   = idx / (512 * ktiles);
    int n    = nt * 16 + (lane & 15);
    int base = kt * 32 + ((lane >> 4) & 1) * 8;
    int k    = base + (s < 8 ? s : s + 8);
    P[idx] = (_Float16)W[k * Nn + n];
}

// ---- layer 1 dense + fused self-loop/bias init -----------------------------
__global__ void k_lin4_fused(const float* __restrict__ x, const float* __restrict__ W1,
                             const float* __restrict__ dinv, const float* __restrict__ b,
                             float* __restrict__ hw, float* __restrict__ acc, int n) {
    int idx = blockIdx.x * blockDim.x + threadIdx.x;
    if (idx >= n * HID) return;
    int i = idx >> 6, j = idx & 63;
    const float* xr = x + (size_t)i * 4;
    float s = xr[0] * W1[j] + xr[1] * W1[64 + j] + xr[2] * W1[128 + j] + xr[3] * W1[192 + j];
    hw[idx] = s;
    float dv = dinv[i];
    acc[idx] = s * dv * dv + b[j];
}

// ---- WMMA GEMM + fused epilogue: hw = A@W ; acc = hw*dinv^2 + b ------------
// one wave per (mtile, ntile); ktiles = 2; N % 16 == 0 so EXEC is all-ones.
__global__ void __launch_bounds__(256) k_gemm64_fused(const _Float16* __restrict__ A16,
                                                      const _Float16* __restrict__ Wp,
                                                      const float* __restrict__ dinv,
                                                      const float* __restrict__ bias,
                                                      float* __restrict__ hw,
                                                      float* __restrict__ acc,
                                                      int mtiles) {
    int wave = (blockIdx.x * blockDim.x + threadIdx.x) >> 5;
    int lane = threadIdx.x & 31;
    int nt = wave & 3;
    int mt = wave >> 2;
    if (mt >= mtiles) return;          // wave-uniform exit, EXEC stays full
    int sel = lane >> 4;               // 0: K {0-7,16-23}, 1: K {8-15,24-31}
    int r   = lane & 15;

    v8f c = {};
#pragma unroll
    for (int kt = 0; kt < 2; ++kt) {
        const uint4* pa = (const uint4*)(A16 + ((size_t)(mt * 16 + r)) * HID + kt * 32 + sel * 8);
        V16HU ua, ub;
        ua.q[0] = pa[0];               // halves base+0..7
        ua.q[1] = pa[2];               // halves base+16..23
        const uint4* pb = (const uint4*)(Wp + ((size_t)((nt * 2 + kt) * 32 + lane)) * 16);
        ub.q[0] = pb[0];
        ub.q[1] = pb[1];
        c = __builtin_amdgcn_wmma_f32_16x16x32_f16(
            /*neg_a=*/false, ua.v, /*neg_b=*/false, ub.v,
            /*c_mod=*/(short)0, c, /*reuse_a=*/false, /*reuse_b=*/false);
    }
    // D layout: VGPR rr -> row mt*16 + sel*8 + rr, col = nt*16 + (lane&15)
    int col     = nt * 16 + r;
    int rowbase = mt * 16 + sel * 8;
    float bj    = bias[col];
#pragma unroll
    for (int rr = 0; rr < 8; ++rr) {
        int row  = rowbase + rr;
        float d  = c[rr];
        float dv = dinv[row];
        hw [(size_t)row * HID + col] = d;
        acc[(size_t)row * HID + col] = d * dv * dv + bj;
    }
}

// ---- edge scatter: acc[dst] += hw[src] * norm (16 threads/edge, b128) ------
__global__ void k_scatter4(const int* __restrict__ src, const int* __restrict__ dst,
                           const float* __restrict__ norm, const float* __restrict__ hw,
                           float* __restrict__ acc, int e) {
    int t   = blockIdx.x * blockDim.x + threadIdx.x;
    int eid = t >> 4;
    int f4  = (t & 15) << 2;
    if (eid >= e) return;
    int s = src[eid], d = dst[eid];
    float nm = norm[eid];
    const float4 v = *(const float4*)(hw + (size_t)s * HID + f4);
    float* a = acc + (size_t)d * HID + f4;
    atomicAdd(a + 0, v.x * nm);
    atomicAdd(a + 1, v.y * nm);
    atomicAdd(a + 2, v.z * nm);
    atomicAdd(a + 3, v.w * nm);
}

// ---- ReLU + f16 convert (4 elements/thread: b128 in, b64 out) --------------
__global__ void k_relu_f16v(const float* __restrict__ acc, _Float16* __restrict__ h16, int n4) {
    int idx = blockIdx.x * blockDim.x + threadIdx.x;
    if (idx >= n4) return;
    float4 v = ((const float4*)acc)[idx];
    H4U o;
    o.h[0] = (_Float16)(v.x > 0.0f ? v.x : 0.0f);
    o.h[1] = (_Float16)(v.y > 0.0f ? v.y : 0.0f);
    o.h[2] = (_Float16)(v.z > 0.0f ? v.z : 0.0f);
    o.h[3] = (_Float16)(v.w > 0.0f ? v.w : 0.0f);
    ((uint2*)h16)[idx] = o.u;
}

// ---- graph mean-pool (16 threads/node, b128 loads) + head ------------------
__global__ void k_pool4(const int* __restrict__ batch, const float* __restrict__ h3,
                        float* __restrict__ sums, float* __restrict__ cnts, int n) {
    int t  = blockIdx.x * blockDim.x + threadIdx.x;
    int i  = t >> 4;
    int f4 = (t & 15) << 2;
    if (i >= n) return;
    int g = batch[i];
    const float4 v = *(const float4*)(h3 + (size_t)i * HID + f4);
    float* sg = sums + (size_t)g * HID + f4;
    atomicAdd(sg + 0, v.x);
    atomicAdd(sg + 1, v.y);
    atomicAdd(sg + 2, v.z);
    atomicAdd(sg + 3, v.w);
    if (f4 == 0) atomicAdd(&cnts[g], 1.0f);
}

__global__ void k_head(const float* __restrict__ sums, const float* __restrict__ cnts,
                       const float* __restrict__ Wpre, const float* __restrict__ bpre,
                       const float* __restrict__ Wlin, const float* __restrict__ blin,
                       float* __restrict__ out) {
    int t = threadIdx.x;          // 256 threads: 64 graphs x 4 output columns
    int g = t >> 2, c = t & 3;
    float inv = 1.0f / fmaxf(cnts[g], 1.0f);
    float r = 0.0f;
    for (int j = 0; j < 32; ++j) {
        float p = bpre[j];
        for (int k = 0; k < HID; ++k) p += sums[g * HID + k] * inv * Wpre[k * 32 + j];
        r += p * Wlin[j * 4 + c];
    }
    out[t] = r + blin[c];
}

// ---------------------------------------------------------------------------
extern "C" void kernel_launch(void* const* d_in, const int* in_sizes, int n_in,
                              void* d_out, int out_size, void* d_ws, size_t ws_size,
                              hipStream_t stream) {
    const float* x     = (const float*)d_in[0];
    const int*   eidx  = (const int*)d_in[1];
    const int*   batch = (const int*)d_in[2];
    const float* W1    = (const float*)d_in[3];
    const float* b1    = (const float*)d_in[4];
    const float* W2    = (const float*)d_in[5];
    const float* b2    = (const float*)d_in[6];
    const float* W3    = (const float*)d_in[7];
    const float* b3    = (const float*)d_in[8];
    const float* Wpre  = (const float*)d_in[9];
    const float* bpre  = (const float*)d_in[10];
    const float* Wlin  = (const float*)d_in[11];
    const float* blin  = (const float*)d_in[12];
    float* out = (float*)d_out;

    const int N = in_sizes[0] / 4;        // 50000 (divisible by 16)
    const int E = in_sizes[1] / 2;        // 800000
    const int* src = eidx;
    const int* dst = eidx + E;

    // ---- workspace carving (all sub-buffers 16B aligned) ----
    float* ws   = (float*)d_ws;
    float* dinv = ws;                               // N
    float* norm = dinv + N;                         // E
    float* hw   = norm + E;                         // N*64
    float* acc  = hw + (size_t)N * HID;             // N*64  (also final h3)
    float* sums = acc + (size_t)N * HID;            // 64*64
    float* cnts = sums + 64 * HID;                  // 64
    _Float16* h16 = (_Float16*)(cnts + 64);         // N*64 halves
    _Float16* W2p = h16 + (size_t)N * HID;          // 4096 halves
    _Float16* W3p = W2p + 4096;                     // 4096 halves

    const int NH   = N * HID;
    const int B    = 256;
    const int gN   = (N + B - 1) / B;
    const int gE   = (E + B - 1) / B;
    const int gNH  = (NH + B - 1) / B;
    const int gE4  = (int)(((size_t)E * 16 + B - 1) / B);   // 16 threads/edge
    const int gN4  = (int)(((size_t)N * 16 + B - 1) / B);   // 16 threads/node
    const int gV4  = (NH / 4 + B - 1) / B;                  // 4 elems/thread
    const int mtiles = N / 16;                              // 3125, exact
    const int gGEMM  = (mtiles * 4 + 7) / 8;                // 8 waves / block

    // ---- normalization ----
    k_fill<<<gN, B, 0, stream>>>(dinv, 1.0f, N);            // self-loop degree
    k_deg_acc<<<gE, B, 0, stream>>>(dst, dinv, E);
    k_rsqrt_inplace<<<gN, B, 0, stream>>>(dinv, N);
    k_edge_norm<<<gE, B, 0, stream>>>(src, dst, dinv, norm, E);

    // ---- pack weights for WMMA (64x64 -> 4 ntiles x 2 ktiles) ----
    k_pack_w<<<(4096 + B - 1) / B, B, 0, stream>>>(W2, W2p, HID, HID);
    k_pack_w<<<(4096 + B - 1) / B, B, 0, stream>>>(W3, W3p, HID, HID);

    // ---- layer 1: dense (K=4) + fused init, scatter, relu ----
    k_lin4_fused<<<gNH, B, 0, stream>>>(x, W1, dinv, b1, hw, acc, N);
    k_scatter4<<<gE4, B, 0, stream>>>(src, dst, norm, hw, acc, E);
    k_relu_f16v<<<gV4, B, 0, stream>>>(acc, h16, NH / 4);

    // ---- layer 2: WMMA GEMM + fused init, scatter, relu ----
    k_gemm64_fused<<<gGEMM, B, 0, stream>>>(h16, W2p, dinv, b2, hw, acc, mtiles);
    k_scatter4<<<gE4, B, 0, stream>>>(src, dst, norm, hw, acc, E);
    k_relu_f16v<<<gV4, B, 0, stream>>>(acc, h16, NH / 4);

    // ---- layer 3: WMMA GEMM + fused init, scatter (no relu; acc == h3) ----
    k_gemm64_fused<<<gGEMM, B, 0, stream>>>(h16, W3p, dinv, b3, hw, acc, mtiles);
    k_scatter4<<<gE4, B, 0, stream>>>(src, dst, norm, hw, acc, E);

    // ---- mean pool + head ----
    k_fill<<<(64 * HID + 64 + B - 1) / B, B, 0, stream>>>(sums, 0.0f, 64 * HID + 64);
    k_pool4<<<gN4, B, 0, stream>>>(batch, acc, sums, cnts, N);
    k_head<<<1, 256, 0, stream>>>(sums, cnts, Wpre, bpre, Wlin, blin, out);
}
